// MultiHeadAttentionLayer_37778532335765
// MI455X (gfx1250) — matrix operand order repaired and checked by reading, not compiled, not run on hardware
//
#include <hip/hip_runtime.h>
#include <hip/hip_bf16.h>

typedef __attribute__((ext_vector_type(16))) _Float16 v16h;
typedef __attribute__((ext_vector_type(8)))  float    v8f;
typedef _Float16 half_t;

static __device__ __forceinline__ v8f wmma_f16(v16h a, v16h b, v8f c) {
  // D = A(16x32 f16) * B(32x16 f16) + C(16x16 f32)
  return __builtin_amdgcn_wmma_f32_16x16x32_f16(
      /*neg_a=*/false, a, /*neg_b=*/false, b,
      /*c_mod=*/(short)0, c, /*reuse_a=*/false, /*reuse_b=*/false);
}

// A fragment from fp32 row (converted), contiguous k = 16*hi + e
static __device__ __forceinline__ void load_a_f32(v16h& a, const float* xr, int hi) {
#pragma unroll
  for (int e = 0; e < 16; ++e) {
    a[e] = (half_t)xr[hi * 16 + e];
  }
}
// A fragment from f16 row
static __device__ __forceinline__ void load_a_f16(v16h& a, const half_t* xr, int hi) {
#pragma unroll
  for (int e = 0; e < 16; ++e) {
    a[e] = xr[hi * 16 + e];
  }
}
// B fragment: 16 contiguous halves per lane (weights stored [n][k])
static __device__ __forceinline__ void load_b_f16(v16h& b, const half_t* wr) {
#pragma unroll
  for (int e = 0; e < 16; ++e) b[e] = wr[e];
}

// ---------------------------------------------------------------------------
// TDM: issue a 2D f16 tile load Global -> LDS via Tensor Data Mover.
// D# layout per CDNA5 ISA ch.8 (group0 128b, group1 256b, groups2/3 zero = 2D).
// 6-arg builtin: (u32x4 g0, i32x8 g1, i32x4 g2, i32x4 g3, i32x8 extra, i32 cpol)
// ---------------------------------------------------------------------------
#if defined(__HIP_DEVICE_COMPILE__) && __has_builtin(__builtin_amdgcn_tensor_load_to_lds)
#define USE_TDM 1
#else
#define USE_TDM 0
#endif

#if USE_TDM
typedef __attribute__((ext_vector_type(4))) unsigned int u32x4;
typedef __attribute__((ext_vector_type(8))) int i32x8;
typedef __attribute__((ext_vector_type(4))) int i32x4;

static __device__ __forceinline__ void tdm_load_2d_f16(
    unsigned lds_addr, const void* gptr,
    unsigned tile_d0, unsigned tile_d1,
    unsigned long long tensor_d0, unsigned long long tensor_d1,
    unsigned long long stride0 /* elements between dim1 lines */) {
  union { unsigned long long q[2]; u32x4 v; } g0;
  union { unsigned long long q[4]; i32x8 v; } g1;
  unsigned long long ga = (unsigned long long)(uintptr_t)gptr;
  // group0: [1:0]count=1 | [63:32]lds_addr | [120:64]global_addr | [127:126]type=2
  g0.q[0] = 1ull | ((unsigned long long)lds_addr << 32);
  g0.q[1] = (ga & 0x01FFFFFFFFFFFFFFull) | (2ull << 62);
  // group1: [15:0]wg_mask=0 | [17:16]data_size=1(2B) | [79:48]tensor_dim0
  //         [111:80]tensor_dim1 | [127:112]tile_dim0 | [143:128]tile_dim1
  //         [159:144]tile_dim2=0 | [207:160]tensor_dim0_stride | [255:208]stride1=0
  g1.q[0] = (1ull << 16) | ((tensor_d0 & 0xFFFFull) << 48);
  g1.q[1] = ((tensor_d0 >> 16) & 0xFFFFull)
          | ((tensor_d1 & 0xFFFFFFFFull) << 16)
          | ((unsigned long long)tile_d0 << 48);
  g1.q[2] = (unsigned long long)tile_d1 | ((stride0 & 0xFFFFFFFFull) << 32);
  g1.q[3] = (stride0 >> 32) & 0xFFFFull;
  i32x4 z4 = {0, 0, 0, 0};
  i32x8 z8 = {0, 0, 0, 0, 0, 0, 0, 0};
  __builtin_amdgcn_tensor_load_to_lds(g0.v, g1.v, z4, z4, z8, 0);
}
#endif

// ---------------------------------------------------------------------------
// Stage 0: weight convert fp32 -> f16 with transpose: dst[b][n][k] = src[b][k][n]
// ---------------------------------------------------------------------------
__global__ void cvt_transpose_kernel(const float* __restrict__ src,
                                     half_t* __restrict__ dst,
                                     int B, int K, int N) {
  long i = (long)blockIdx.x * blockDim.x + threadIdx.x;
  long total = (long)B * K * N;
  if (i >= total) return;
  int k = (int)(i % K);
  long t = i / K;
  int n = (int)(t % N);
  int b = (int)(t / N);
  dst[i] = (half_t)src[((long)b * K + k) * N + n];
}

// ---------------------------------------------------------------------------
// Stage 1: per-head projection  Out[h] = X (4096x512 fp32) @ W[h] (512x64)
// WT layout: (H, 64, 512) f16 [n][k].  Out row-major (H,4096,64) or (H,64,4096).
// grid = (4096/128, H), block = 256 (8 waves); wave owns a 16x64 output tile.
// Software-pipelined: ping-pong fragment sets so loads for slab k+32 overlap
// the WMMA chain consuming slab k.
// ---------------------------------------------------------------------------
__global__ void __launch_bounds__(256)
proj_gemm_kernel(const float* __restrict__ X,
                 const half_t* __restrict__ WT,
                 half_t* __restrict__ Out,
                 int transpose_out) {
  const int NROW = 4096, DIM = 512, D = 64;
  const int h    = blockIdx.y;
  const int wave = threadIdx.x >> 5;
  const int lane = threadIdx.x & 31;
  const int m    = lane & 15;
  const int hi   = lane >> 4;
  const int row0 = blockIdx.x * 128 + wave * 16;
  const half_t* wt = WT + (long)h * D * DIM;
  const float*  xb = X + (long)(row0 + m) * DIM;

  v8f acc[4] = {};
  v16h aA, aB, bA[4], bB[4];

  load_a_f32(aA, xb, hi);
#pragma unroll
  for (int nt = 0; nt < 4; ++nt)
    load_b_f16(bA[nt], wt + (long)(nt * 16 + m) * DIM + hi * 16);

  for (int k0 = 0; k0 < DIM; k0 += 64) {
    const int k1 = k0 + 32;
    load_a_f32(aB, xb + k1, hi);
#pragma unroll
    for (int nt = 0; nt < 4; ++nt)
      load_b_f16(bB[nt], wt + (long)(nt * 16 + m) * DIM + k1 + hi * 16);
#pragma unroll
    for (int nt = 0; nt < 4; ++nt) acc[nt] = wmma_f16(aA, bA[nt], acc[nt]);

    const int k2 = (k0 + 64) & (DIM - 1);   // branchless wrap on last iter
    load_a_f32(aA, xb + k2, hi);
#pragma unroll
    for (int nt = 0; nt < 4; ++nt)
      load_b_f16(bA[nt], wt + (long)(nt * 16 + m) * DIM + k2 + hi * 16);
#pragma unroll
    for (int nt = 0; nt < 4; ++nt) acc[nt] = wmma_f16(aB, bB[nt], acc[nt]);
  }

  if (!transpose_out) {
#pragma unroll
    for (int nt = 0; nt < 4; ++nt)
#pragma unroll
      for (int r = 0; r < 8; ++r) {
        int row = row0 + hi * 8 + r;              // C layout: M = 8*hi + r
        Out[((long)h * NROW + row) * D + nt * 16 + m] = (half_t)acc[nt][r];
      }
  } else {
#pragma unroll
    for (int nt = 0; nt < 4; ++nt) {
      half_t* o = Out + ((long)h * D + nt * 16 + m) * NROW + row0 + hi * 8;
#pragma unroll
      for (int r = 0; r < 8; ++r) o[r] = (half_t)acc[nt][r];
    }
  }
}

// ---------------------------------------------------------------------------
// Stage 2: flash attention per head, K/V tiles staged in LDS (double-buffered).
// Qh,Kh: (H,4096,64) f16.  VTh: (H,64,4096) f16 (transposed).
// O: (4096,512) f16 head-concat.  grid = (4096/128, H), block = 256.
// Wave 0 drives the Tensor Data Mover; one barrier per 32-key block overlaps
// the DMA of tile j+1 with the 8 WMMAs + softmax of tile j.
// ---------------------------------------------------------------------------
__global__ void __launch_bounds__(256)
flash_attn_kernel(const half_t* __restrict__ Qh,
                  const half_t* __restrict__ Kh,
                  const half_t* __restrict__ VTh,
                  half_t* __restrict__ O) {
  const int NKEY = 4096, D = 64;
  const float scale = 0.125f; // 1/sqrt(64)

  __shared__ half_t kbuf[2][32 * 64];   // 2 x 4 KB: 32 keys x 64 feats
  __shared__ half_t vbuf[2][64 * 32];   // 2 x 4 KB: 64 feats x 32 keys (V^T)
  __shared__ half_t pbuf[8][16 * 32];   // per-wave P transpose tile (8 KB)

  const int h    = blockIdx.y;
  const int wave = threadIdx.x >> 5;
  const int lane = threadIdx.x & 31;
  const int m    = lane & 15;
  const int hi   = lane >> 4;
  const int row0 = blockIdx.x * 128 + wave * 16;
  const bool w0  = (threadIdx.x < 32);

  const half_t* kp = Kh + (long)h * NKEY * D;
  const half_t* vt = VTh + (long)h * D * NKEY;

  // Q tile 16x64 in registers: two A fragments (K chunks of 32)
  v16h aq0, aq1;
  {
    const half_t* qr = Qh + ((long)h * NKEY + row0 + m) * D;
    load_a_f16(aq0, qr, hi);
    load_a_f16(aq1, qr + 32, hi);
  }

  float mr[8], lr[8], alpha[8];
  v8f acc[4] = {};
#pragma unroll
  for (int r = 0; r < 8; ++r) { mr[r] = -__builtin_inff(); lr[r] = 0.0f; }

  // ---- prologue: stage tile j=0 into buffer 0 ----
#if USE_TDM
  if (w0) {
    tdm_load_2d_f16((unsigned)(uintptr_t)&kbuf[0][0], kp,
                    /*tile*/ 64, 32, /*tensor*/ 64, NKEY, /*stride0*/ 64);
    tdm_load_2d_f16((unsigned)(uintptr_t)&vbuf[0][0], vt,
                    /*tile*/ 32, 64, /*tensor*/ NKEY, D, /*stride0*/ NKEY);
  }
#else
  {
    const int vr0 = threadIdx.x >> 2, vc = threadIdx.x & 3;  // V: 4 thr/row
    uint4 nk = *(const uint4*)((const char*)kp + threadIdx.x * 16);
    uint4 nv = *(const uint4*)((const char*)(vt + (long)vr0 * NKEY) + vc * 16);
    *(uint4*)((char*)&kbuf[0][0] + threadIdx.x * 16) = nk;
    *(uint4*)((char*)&vbuf[0][0] + vr0 * 64 + vc * 16) = nv;
  }
#endif

  for (int jt = 0; jt < NKEY / 32; ++jt) {
    const int j = jt * 32;
    const int cur = jt & 1;

#if USE_TDM
    if (w0) __builtin_amdgcn_s_wait_tensorcnt(0);  // current buffer landed
    __syncthreads();                               // publish; prev buf free
    if (w0 && jt + 1 < NKEY / 32) {                // kick DMA for next tile
      const int jn = j + 32;
      tdm_load_2d_f16((unsigned)(uintptr_t)&kbuf[cur ^ 1][0],
                      kp + (long)jn * D, 64, 32, 64, NKEY, 64);
      tdm_load_2d_f16((unsigned)(uintptr_t)&vbuf[cur ^ 1][0],
                      vt + jn, 32, 64, NKEY, D, NKEY);
    }
#else
    __syncthreads();
    uint4 nk, nv;
    const int vr0 = threadIdx.x >> 2, vc = threadIdx.x & 3;
    const bool have_next = (jt + 1 < NKEY / 32);
    if (have_next) {
      const int jn = j + 32;
      nk = *(const uint4*)((const char*)(kp + (long)jn * D) + threadIdx.x * 16);
      nv = *(const uint4*)((const char*)(vt + (long)vr0 * NKEY + jn) + vc * 16);
    }
#endif

    const half_t* kb = kbuf[cur];
    const half_t* vb = vbuf[cur];

    // ---- scores: load all 4 K fragments from LDS, then 4 WMMAs ----
    v8f s0 = {}, s1 = {};
    {
      v16h bk[4];
#pragma unroll
      for (int t = 0; t < 2; ++t) {        // key sub-tile (j..j+15, j+16..j+31)
        const half_t* kr = kb + (t * 16 + m) * 64 + hi * 16;
        load_b_f16(bk[t * 2 + 0], kr);       // feature chunk 0
        load_b_f16(bk[t * 2 + 1], kr + 32);  // feature chunk 1
      }
      s0 = wmma_f16(aq0, bk[0], s0);
      s0 = wmma_f16(aq1, bk[1], s0);
      s1 = wmma_f16(aq0, bk[2], s1);
      s1 = wmma_f16(aq1, bk[3], s1);
    }

    // ---- online softmax (row = 8*hi + r, 16 cols across the lane half) ----
#pragma unroll
    for (int r = 0; r < 8; ++r) {
      float t0 = s0[r] * scale, t1 = s1[r] * scale;
      float mx = fmaxf(t0, t1);
      mx = fmaxf(mx, __shfl_xor(mx, 1, 32));
      mx = fmaxf(mx, __shfl_xor(mx, 2, 32));
      mx = fmaxf(mx, __shfl_xor(mx, 4, 32));
      mx = fmaxf(mx, __shfl_xor(mx, 8, 32));
      float mn = fmaxf(mr[r], mx);
      float a  = __expf(mr[r] - mn);
      float p0 = __expf(t0 - mn);
      float p1 = __expf(t1 - mn);
      float rs = p0 + p1;
      rs += __shfl_xor(rs, 1, 32);
      rs += __shfl_xor(rs, 2, 32);
      rs += __shfl_xor(rs, 4, 32);
      rs += __shfl_xor(rs, 8, 32);
      lr[r] = lr[r] * a + rs;
      mr[r] = mn;
      alpha[r] = a;
      s0[r] = p0;
      s1[r] = p1;
    }
#pragma unroll
    for (int nt = 0; nt < 4; ++nt)
#pragma unroll
      for (int r = 0; r < 8; ++r) acc[nt][r] *= alpha[r];

    // ---- transpose P (C layout) -> A layout via per-wave LDS tile ----
    half_t* pb = pbuf[wave];
#pragma unroll
    for (int r = 0; r < 8; ++r) {
      int row = hi * 8 + r;
      pb[row * 32 + m]      = (half_t)s0[r];
      pb[row * 32 + 16 + m] = (half_t)s1[r];
    }
    v16h ap;
    load_a_f16(ap, pb + m * 32, hi);   // same-wave DS ops are in-order

    // ---- O += P @ V : load all 4 V fragments from LDS, then 4 WMMAs ----
    {
      v16h bv[4];
#pragma unroll
      for (int nt = 0; nt < 4; ++nt)
        load_b_f16(bv[nt], vb + (nt * 16 + m) * 32 + hi * 16);
#pragma unroll
      for (int nt = 0; nt < 4; ++nt) acc[nt] = wmma_f16(ap, bv[nt], acc[nt]);
    }

#if !USE_TDM
    if (have_next) {
      *(uint4*)((char*)&kbuf[cur ^ 1][0] + threadIdx.x * 16) = nk;
      *(uint4*)((char*)&vbuf[cur ^ 1][0] + vr0 * 64 + vc * 16) = nv;
    }
#endif
  }

  // ---- normalize and store head-concat output ----
#pragma unroll
  for (int r = 0; r < 8; ++r) lr[r] = 1.0f / lr[r];
#pragma unroll
  for (int nt = 0; nt < 4; ++nt)
#pragma unroll
    for (int r = 0; r < 8; ++r) {
      int row = row0 + hi * 8 + r;
      O[(long)row * 512 + h * 64 + nt * 16 + m] = (half_t)(acc[nt][r] * lr[r]);
    }
}

// ---------------------------------------------------------------------------
// Stage 3: out = O (4096x512 f16) @ Wo (512x512) -> fp32.  WoT: (512,512) [n][k].
// grid = (4096/128, 512/64), block = 256.  Same ping-pong pipeline.
// ---------------------------------------------------------------------------
__global__ void __launch_bounds__(256)
out_gemm_kernel(const half_t* __restrict__ A,
                const half_t* __restrict__ WoT,
                float* __restrict__ Out) {
  const int DIM = 512;
  const int wave = threadIdx.x >> 5;
  const int lane = threadIdx.x & 31;
  const int m    = lane & 15;
  const int hi   = lane >> 4;
  const int row0 = blockIdx.x * 128 + wave * 16;
  const int col0 = blockIdx.y * 64;
  const half_t* ab = A + (long)(row0 + m) * DIM;

  v8f acc[4] = {};
  v16h aA, aB, bA[4], bB[4];

  load_a_f16(aA, ab, hi);
#pragma unroll
  for (int nt = 0; nt < 4; ++nt)
    load_b_f16(bA[nt], WoT + (long)(col0 + nt * 16 + m) * DIM + hi * 16);

  for (int k0 = 0; k0 < DIM; k0 += 64) {
    const int k1 = k0 + 32;
    load_a_f16(aB, ab + k1, hi);
#pragma unroll
    for (int nt = 0; nt < 4; ++nt)
      load_b_f16(bB[nt], WoT + (long)(col0 + nt * 16 + m) * DIM + k1 + hi * 16);
#pragma unroll
    for (int nt = 0; nt < 4; ++nt) acc[nt] = wmma_f16(aA, bA[nt], acc[nt]);

    const int k2 = (k0 + 64) & (DIM - 1);   // branchless wrap on last iter
    load_a_f16(aA, ab + k2, hi);
#pragma unroll
    for (int nt = 0; nt < 4; ++nt)
      load_b_f16(bA[nt], WoT + (long)(col0 + nt * 16 + m) * DIM + k2 + hi * 16);
#pragma unroll
    for (int nt = 0; nt < 4; ++nt) acc[nt] = wmma_f16(aB, bB[nt], acc[nt]);
  }
#pragma unroll
  for (int nt = 0; nt < 4; ++nt)
#pragma unroll
    for (int r = 0; r < 8; ++r)
      Out[(long)(row0 + hi * 8 + r) * DIM + col0 + nt * 16 + m] = acc[nt][r];
}

// ---------------------------------------------------------------------------
extern "C" void kernel_launch(void* const* d_in, const int* in_sizes, int n_in,
                              void* d_out, int out_size, void* d_ws, size_t ws_size,
                              hipStream_t stream) {
  const float* Q  = (const float*)d_in[0];
  const float* K  = (const float*)d_in[1];
  const float* V  = (const float*)d_in[2];
  const float* Wq = (const float*)d_in[3];
  const float* Wk = (const float*)d_in[4];
  const float* Wv = (const float*)d_in[5];
  const float* Wo = (const float*)d_in[6];
  float* out = (float*)d_out;

  const int N = 4096, DIM = 512, H = 8, D = 64;

  // workspace carve-up (f16 elements), total ~18 MB
  half_t* ws  = (half_t*)d_ws;
  half_t* wqT = ws;                              // H*D*DIM
  half_t* wkT = wqT + (size_t)H * D * DIM;
  half_t* wvT = wkT + (size_t)H * D * DIM;
  half_t* woT = wvT + (size_t)H * D * DIM;       // DIM*DIM
  half_t* qh  = woT + (size_t)DIM * DIM;         // H*N*D
  half_t* kh  = qh  + (size_t)H * N * D;
  half_t* vth = kh  + (size_t)H * N * D;         // transposed V: (H, D, N)
  half_t* oh  = vth + (size_t)H * N * D;         // N*DIM head-concat attn out

  // Stage 0: weight convert+transpose
  {
    int t1 = H * DIM * D;
    cvt_transpose_kernel<<<(t1 + 255) / 256, 256, 0, stream>>>(Wq, wqT, H, DIM, D);
    cvt_transpose_kernel<<<(t1 + 255) / 256, 256, 0, stream>>>(Wk, wkT, H, DIM, D);
    cvt_transpose_kernel<<<(t1 + 255) / 256, 256, 0, stream>>>(Wv, wvT, H, DIM, D);
    int t2 = DIM * DIM;
    cvt_transpose_kernel<<<(t2 + 255) / 256, 256, 0, stream>>>(Wo, woT, 1, DIM, DIM);
  }

  // Stage 1: q/k/v projections (V stored transposed for P@V fragment loads)
  dim3 gproj(N / 128, H);
  proj_gemm_kernel<<<gproj, 256, 0, stream>>>(Q, wqT, qh, 0);
  proj_gemm_kernel<<<gproj, 256, 0, stream>>>(K, wkT, kh, 0);
  proj_gemm_kernel<<<gproj, 256, 0, stream>>>(V, wvT, vth, 1);

  // Stage 2: flash attention (scores never touch HBM; TDM streams K/V tiles)
  flash_attn_kernel<<<gproj, 256, 0, stream>>>(qh, kh, vth, oh);

  // Stage 3: output projection
  dim3 gout(N / 128, DIM / 64);
  out_gemm_kernel<<<gout, 256, 0, stream>>>(oh, woT, out);
}
